// AAREN_35450660061986
// MI455X (gfx1250) — matrix operand order repaired and checked by reading, not compiled, not run on hardware
//
#include <hip/hip_runtime.h>
#include <math.h>

#define B_  4
#define T_  4096
#define D_  128
#define H_  8
#define E_  (H_*D_*2)      // 2048
#define M_  (B_*T_)        // 16384
#define NCHUNK 16
#define CLEN (T_/NCHUNK)   // 256

typedef __attribute__((ext_vector_type(16))) __bf16 v16bf;
typedef __attribute__((ext_vector_type(8)))  float  v8f;

__device__ __forceinline__ unsigned short f32_to_bf16_rne(float f) {
  unsigned u = __float_as_uint(f);
  u += 0x7FFFu + ((u >> 16) & 1u);
  return (unsigned short)(u >> 16);
}

// ---------------- K0: fp32 -> bf16 convert (x and W_kv) ----------------
__global__ void aaren_convert(const float* __restrict__ x, const float* __restrict__ wkv,
                              unsigned short* __restrict__ xbf, unsigned short* __restrict__ wbf) {
  int i = blockIdx.x * blockDim.x + threadIdx.x;
  const int nx = M_ * D_;
  const int nw = E_ * D_;
  if (i < nx)            xbf[i]      = f32_to_bf16_rne(x[i]);
  else if (i < nx + nw)  wbf[i - nx] = f32_to_bf16_rne(wkv[i - nx]);
}

__global__ void aaren_zero(float* __restrict__ out, int n) {
  int i = blockIdx.x * blockDim.x + threadIdx.x;
  if (i < n) out[i] = 0.0f;
}

// ---------------- K1: fused bf16 WMMA kv-GEMM + SiLU + score ----------------
// grid: (M/64, H), block: 256 (8 wave32). Each block: 64 rows x one full head (256 e-cols).
__global__ __launch_bounds__(256) void aaren_kvgemm(
    const unsigned short* __restrict__ xbf,
    const unsigned short* __restrict__ wbf,
    const float* __restrict__ qproj,
    float* __restrict__ v_ws, float* __restrict__ s_ws) {
  __shared__ unsigned short Xs [64][136];   // padded rows: conflict-free ds reads
  __shared__ unsigned short Wsh[256][136];
  __shared__ float qs[D_];
  __shared__ float s_part[4][2][16];

  const int tid = threadIdx.x;
  const int h   = blockIdx.y;
  const int m0  = blockIdx.x * 64;
  const int e0  = h * 256;

  // stage X tile (64x128 bf16) with 16B chunks
  for (int idx = tid; idx < 64 * 16; idx += 256) {
    int r = idx >> 4, c8 = (idx & 15) << 3;
    *(uint4*)&Xs[r][c8] = *(const uint4*)&xbf[(size_t)(m0 + r) * D_ + c8];
  }
  // stage W tile (256x128 bf16): B[k,n] = W_kv[e0+n, k] (rows contiguous in K)
  for (int idx = tid; idx < 256 * 16; idx += 256) {
    int r = idx >> 4, c8 = (idx & 15) << 3;
    *(uint4*)&Wsh[r][c8] = *(const uint4*)&wbf[(size_t)(e0 + r) * D_ + c8];
  }
  if (tid < D_) qs[tid] = qproj[h * D_ + tid];
  __syncthreads();

  const int lane  = tid & 31;
  const int wid   = tid >> 5;
  const int mtile = wid >> 1;   // 0..3 -> 16 M rows each
  const int nhalf = wid & 1;    // 0..1 -> 128 e-cols each
  const int l16   = lane & 15;
  const bool hi   = (lane >= 16);

  v8f zero8 = {0.f,0.f,0.f,0.f,0.f,0.f,0.f,0.f};
  v8f acc[8];
  #pragma unroll
  for (int i = 0; i < 8; ++i) acc[i] = zero8;

  #pragma unroll
  for (int kc = 0; kc < 4; ++kc) {            // K = 128 in 4 steps of 32
    const int K0 = kc * 32;
    // A fragment, 16x32 bf16, ISA 7.12.2 layout (lane row = lane%16)
    union { unsigned u[8]; v16bf v; } af;
    const int ar = mtile * 16 + l16;
    const int kb = hi ? 8 : 0;
    #pragma unroll
    for (int i = 0; i < 8; ++i) {
      int k = kb + ((i < 4) ? (2 * i) : (16 + 2 * (i - 4)));
      af.u[i] = *(const unsigned*)&Xs[ar][K0 + k];
    }
    #pragma unroll
    for (int nt = 0; nt < 8; ++nt) {
      // B fragment, 32x16 bf16: N = lane%16, K in same interleaved order as A
      union { unsigned u[8]; v16bf v; } bfr;
      const int br = nhalf * 128 + nt * 16 + l16;
      const int kh = hi ? 8 : 0;
      #pragma unroll
      for (int j = 0; j < 8; ++j)
        bfr.u[j] = *(const unsigned*)&Wsh[br][K0 + kh + ((j < 4) ? (2 * j) : (16 + 2 * (j - 4)))];
      acc[nt] = __builtin_amdgcn_wmma_f32_16x16x32_bf16(
          false, af.v, false, bfr.v, (short)0, acc[nt], false, false);
    }
  }

  // epilogue: SiLU; even e-cols -> k (score partial), odd -> v (store)
  float part[8];
  #pragma unroll
  for (int j = 0; j < 8; ++j) part[j] = 0.f;

  const int rowoff = hi ? 8 : 0;   // C/D layout: VGPR j -> M=j (lanes 0-15) / j+8 (16-31)
  #pragma unroll
  for (int nt = 0; nt < 8; ++nt) {
    const int ncol = nhalf * 128 + nt * 16 + l16;   // 0..255 within head: e = h*256+ncol
    const int d    = ncol >> 1;
    const bool isV = (ncol & 1);
    #pragma unroll
    for (int j = 0; j < 8; ++j) {
      const int m = m0 + mtile * 16 + j + rowoff;
      float z  = acc[nt][j];
      float sv = z / (1.0f + __expf(-z));           // silu
      if (isV) {
        const int b = m >> 12, t = m & 4095;
        v_ws[((size_t)(b * H_ + h) * T_ + t) * D_ + d] = sv;
      } else {
        part[j] += qs[d] * sv;
      }
    }
  }

  // reduce score partials within each 16-lane half (M differs across halves)
  #pragma unroll
  for (int j = 0; j < 8; ++j) {
    float p = part[j];
    #pragma unroll
    for (int off = 1; off < 16; off <<= 1) p += __shfl_xor(p, off, 32);
    if (l16 == 0) s_part[mtile][nhalf][j + rowoff] = p;
  }
  __syncthreads();
  if (tid < 64) {
    const int mt = tid >> 4, r = tid & 15;
    const int m = m0 + mt * 16 + r;
    const int b = m >> 12, t = m & 4095;
    s_ws[(size_t)(b * H_ + h) * T_ + t] = s_part[mt][0][r] + s_part[mt][1][r];
  }
}

// ---------------- K2: per-chunk scan summaries (M, U, W[128]) ----------------
__global__ __launch_bounds__(128) void aaren_chunk_sum(
    const float* __restrict__ s_ws, const float* __restrict__ v_ws,
    float* __restrict__ Msum, float* __restrict__ Usum, float* __restrict__ Wsum) {
  const int bc    = blockIdx.x;
  const int chunk = bc & (NCHUNK - 1);
  const int row   = bc >> 4;          // row = b*H + h
  const int d     = threadIdx.x;
  const float* sr = s_ws + (size_t)row * T_;
  const float* vr = v_ws + (size_t)row * T_ * D_;
  const int t0 = chunk * CLEN;
  float M = -INFINITY, U = 0.f, W = 0.f;
  float s = sr[t0];
  float v = vr[(size_t)t0 * D_ + d];
  for (int t = t0; t < t0 + CLEN; ++t) {
    float sn = 0.f, vn = 0.f;
    if (t + 1 < t0 + CLEN) { sn = sr[t + 1]; vn = vr[(size_t)(t + 1) * D_ + d]; }
    float Mn = fmaxf(M, s);
    float a  = __expf(M - Mn);
    float e  = __expf(s - Mn);
    U = U * a + e;
    W = W * a + e * v;
    M = Mn;
    s = sn; v = vn;
  }
  if (d == 0) { Msum[bc] = M; Usum[bc] = U; }
  Wsum[(size_t)bc * D_ + d] = W;
}

// ---------------- K3: carry-in fold + replay; head-mean via f32 atomics ----------------
__global__ __launch_bounds__(128) void aaren_apply(
    const float* __restrict__ s_ws, const float* __restrict__ v_ws,
    const float* __restrict__ Msum, const float* __restrict__ Usum,
    const float* __restrict__ Wsum, float* __restrict__ out) {
  const int bc    = blockIdx.x;
  const int chunk = bc & (NCHUNK - 1);
  const int row   = bc >> 4;
  const int d     = threadIdx.x;
  const int b     = row >> 3;
  const float* sr = s_ws + (size_t)row * T_;
  const float* vr = v_ws + (size_t)row * T_ * D_;
  float M = -INFINITY, U = 0.f, W = 0.f;
  const int base = row * NCHUNK;
  for (int c = 0; c < chunk; ++c) {             // exclusive prefix of chunk summaries
    float Mj = Msum[base + c];
    float Uj = Usum[base + c];
    float Wj = Wsum[(size_t)(base + c) * D_ + d];
    float Mn = fmaxf(M, Mj);
    float a1 = __expf(M - Mn), a2 = __expf(Mj - Mn);
    U = U * a1 + Uj * a2;
    W = W * a1 + Wj * a2;
    M = Mn;
  }
  const int t0 = chunk * CLEN;
  for (int t = t0; t < t0 + CLEN; ++t) {
    float s = sr[t];
    float v = vr[(size_t)t * D_ + d];
    float Mn = fmaxf(M, s);
    float a  = __expf(M - Mn);
    float e  = __expf(s - Mn);
    U = U * a + e;
    W = W * a + e * v;
    M = Mn;
    atomicAdd(&out[((size_t)b * T_ + t) * D_ + d], (W / U) * 0.125f);
  }
}

extern "C" void kernel_launch(void* const* d_in, const int* in_sizes, int n_in,
                              void* d_out, int out_size, void* d_ws, size_t ws_size,
                              hipStream_t stream) {
  const float* x   = (const float*)d_in[0];   // [4,4096,128]
  const float* wkv = (const float*)d_in[1];   // [2048,128]
  const float* q   = (const float*)d_in[2];   // [8,128]
  float* out = (float*)d_out;                 // [4,4096,128]

  // workspace layout (~73 MB)
  unsigned short* xbf = (unsigned short*)d_ws;
  unsigned short* wbf = xbf + (size_t)M_ * D_;
  float* vws  = (float*)(wbf + (size_t)E_ * D_);     // byte offset 4.5 MB, 16B aligned
  float* sws  = vws + (size_t)B_ * H_ * T_ * D_;     // 64 MB of v
  float* Msum = sws + (size_t)B_ * H_ * T_;
  float* Usum = Msum + B_ * H_ * NCHUNK;
  float* Wsum = Usum + B_ * H_ * NCHUNK;

  const int nconv = M_ * D_ + E_ * D_;
  aaren_convert<<<(nconv + 255) / 256, 256, 0, stream>>>(x, wkv, xbf, wbf);
  const int nout = B_ * T_ * D_;
  aaren_zero<<<(nout + 255) / 256, 256, 0, stream>>>(out, nout);
  aaren_kvgemm<<<dim3(M_ / 64, H_), 256, 0, stream>>>(xbf, wbf, q, vws, sws);
  aaren_chunk_sum<<<B_ * H_ * NCHUNK, D_, 0, stream>>>(sws, vws, Msum, Usum, Wsum);
  aaren_apply<<<B_ * H_ * NCHUNK, D_, 0, stream>>>(sws, vws, Msum, Usum, Wsum, out);
}
